// MHSABlock_13451837571905
// MI455X (gfx1250) — matrix operand order repaired and checked by reading, not compiled, not run on hardware
//
#include <hip/hip_runtime.h>
#include <hip/hip_bf16.h>
#include <math.h>
#include <stddef.h>

#define BB    8
#define SS    32
#define NHEAD 8
#define NN    (SS*SS + 1)        // 1025
#define CC    512
#define DD    64                 // head dim
#define MROWS (BB*NN)            // 8200
#define NIMG  (NN-1)             // 1024
#define QSCALE 0.125f            // D^-0.5
#define LAMK   0.6f

typedef __attribute__((ext_vector_type(16))) __bf16 v16bf;
typedef __attribute__((ext_vector_type(8)))  __bf16 v8bf;
typedef __attribute__((ext_vector_type(8)))  float  v8f;

static __device__ inline v8f wmma_bf16(v16bf a, v16bf b, v8f c) {
  // D = A(16x32 bf16) x B(32x16 bf16) + C(16x16 f32)
  return __builtin_amdgcn_wmma_f32_16x16x32_bf16(false, a, false, b, (short)0, c,
                                                 false, false);
}

static __device__ inline v16bf zero16() {
  v16bf z;
#pragma unroll
  for (int i = 0; i < 16; ++i) z[i] = (__bf16)0.0f;
  return z;
}

// combine two contiguous 16B halves into one fragment register set
static __device__ inline v16bf loadfrag(const __bf16* p0, const __bf16* p1) {
  v8bf lo = *(const v8bf*)p0;
  v8bf up = *(const v8bf*)p1;
  v16bf r;
#pragma unroll
  for (int i = 0; i < 8; ++i) { r[i] = lo[i]; r[8 + i] = up[i]; }
  return r;
}

// ---------------------------------------------------------------- weight cast
__global__ void cast_f32_bf16(const float* __restrict__ src,
                              __bf16* __restrict__ dst, int n) {
  int i = blockIdx.x * 256 + threadIdx.x;
  if (i < n) dst[i] = (__bf16)src[i];
}

// ---------------------------------------------------------------- layernorm
__global__ __launch_bounds__(256)
void ln_kernel(const float* __restrict__ X, const float* __restrict__ w,
               const float* __restrict__ bias, float* __restrict__ outf,
               __bf16* __restrict__ outb) {
  int r = blockIdx.x, tid = threadIdx.x;
  __shared__ float rs[256], rq[256];
  const float* xr = X + (size_t)r * CC;
  float s = 0.f, q = 0.f;
  for (int c = tid; c < CC; c += 256) { float v = xr[c]; s += v; q += v * v; }
  rs[tid] = s; rq[tid] = q; __syncthreads();
  for (int st = 128; st > 0; st >>= 1) {
    if (tid < st) { rs[tid] += rs[tid + st]; rq[tid] += rq[tid + st]; }
    __syncthreads();
  }
  float mean = rs[0] * (1.0f / CC);
  float var  = rq[0] * (1.0f / CC) - mean * mean;
  float inv  = rsqrtf(var + 1e-5f);
  for (int c = tid; c < CC; c += 256) {
    float y = (xr[c] - mean) * inv * w[c] + bias[c];
    if (outf) outf[(size_t)r * CC + c] = y;
    outb[(size_t)r * CC + c] = (__bf16)y;
  }
}

// ---------------------------------------------------------------- WMMA GEMM
// C[M,Nout] = A[M,K](bf16) @ W[Nout,K]^T(bf16). One wave -> 32x64 tile,
// software-pipelined: next k-step fragments are issued before current WMMAs.
// mode 0: QKV  -> bf16 out, cols < CC (=q) scaled by QSCALE, no bias
// mode 1: PROJ -> f32 out, + bias
// mode 2: FC1  -> bf16 out, + bias, exact GELU
// mode 3: FC2  -> f32 out, + bias + residual
__global__ __launch_bounds__(32)
void gemm_kernel(const __bf16* __restrict__ A, const __bf16* __restrict__ W,
                 const float* __restrict__ bias, const float* __restrict__ resid,
                 float* __restrict__ outf, __bf16* __restrict__ outb,
                 int Mreal, int K, int Nout, int mode) {
  int lane = threadIdx.x;
  int lr = lane & 15, hi = lane >> 4;
  int m0 = blockIdx.x * 32;
  int n0 = blockIdx.y * 64;
  int row0 = m0 + lr, row1 = m0 + 16 + lr;
  bool ok0 = row0 < Mreal, ok1 = row1 < Mreal;
  const __bf16* a0p = A + (size_t)row0 * K + hi * 8;
  const __bf16* a1p = A + (size_t)row1 * K + hi * 8;
  const __bf16* b0p = W + (size_t)(n0 + 0 * 16 + lr) * K + hi * 8;
  const __bf16* b1p = W + (size_t)(n0 + 1 * 16 + lr) * K + hi * 8;
  const __bf16* b2p = W + (size_t)(n0 + 2 * 16 + lr) * K + hi * 8;
  const __bf16* b3p = W + (size_t)(n0 + 3 * 16 + lr) * K + hi * 8;

  v8f acc[8];
#pragma unroll
  for (int t = 0; t < 8; ++t)
#pragma unroll
    for (int i = 0; i < 8; ++i) acc[t][i] = 0.f;

  // prologue: fragments for k0 = 0
  v16bf a0 = ok0 ? loadfrag(a0p, a0p + 16) : zero16();
  v16bf a1 = ok1 ? loadfrag(a1p, a1p + 16) : zero16();
  v16bf bf0 = loadfrag(b0p, b0p + 16);
  v16bf bf1 = loadfrag(b1p, b1p + 16);
  v16bf bf2 = loadfrag(b2p, b2p + 16);
  v16bf bf3 = loadfrag(b3p, b3p + 16);

  for (int k0 = 32; k0 < K; k0 += 32) {
    // issue next k-step loads (stay in flight across the 8 WMMAs below)
    v16bf na0 = ok0 ? loadfrag(a0p + k0, a0p + k0 + 16) : zero16();
    v16bf na1 = ok1 ? loadfrag(a1p + k0, a1p + k0 + 16) : zero16();
    v16bf nb0 = loadfrag(b0p + k0, b0p + k0 + 16);
    v16bf nb1 = loadfrag(b1p + k0, b1p + k0 + 16);
    v16bf nb2 = loadfrag(b2p + k0, b2p + k0 + 16);
    v16bf nb3 = loadfrag(b3p + k0, b3p + k0 + 16);

    acc[0] = wmma_bf16(a0, bf0, acc[0]);
    acc[1] = wmma_bf16(a0, bf1, acc[1]);
    acc[2] = wmma_bf16(a0, bf2, acc[2]);
    acc[3] = wmma_bf16(a0, bf3, acc[3]);
    acc[4] = wmma_bf16(a1, bf0, acc[4]);
    acc[5] = wmma_bf16(a1, bf1, acc[5]);
    acc[6] = wmma_bf16(a1, bf2, acc[6]);
    acc[7] = wmma_bf16(a1, bf3, acc[7]);

    a0 = na0; a1 = na1; bf0 = nb0; bf1 = nb1; bf2 = nb2; bf3 = nb3;
  }
  acc[0] = wmma_bf16(a0, bf0, acc[0]);
  acc[1] = wmma_bf16(a0, bf1, acc[1]);
  acc[2] = wmma_bf16(a0, bf2, acc[2]);
  acc[3] = wmma_bf16(a0, bf3, acc[3]);
  acc[4] = wmma_bf16(a1, bf0, acc[4]);
  acc[5] = wmma_bf16(a1, bf1, acc[5]);
  acc[6] = wmma_bf16(a1, bf2, acc[6]);
  acc[7] = wmma_bf16(a1, bf3, acc[7]);

  for (int mi = 0; mi < 2; ++mi) {
    for (int i = 0; i < 8; ++i) {
      int m = m0 + mi * 16 + i + 8 * hi;
      if (m >= Mreal) continue;
      for (int nb = 0; nb < 4; ++nb) {
        int col = n0 + nb * 16 + lr;
        size_t off = (size_t)m * Nout + col;
        float v = acc[mi * 4 + nb][i];
        if (mode == 0) {
          if (col < CC) v *= QSCALE;
          outb[off] = (__bf16)v;
        } else if (mode == 1) {
          outf[off] = v + bias[col];
        } else if (mode == 2) {
          float t = v + bias[col];
          float g = 0.5f * t * (1.0f + erff(t * 0.70710678118654752f));
          outb[off] = (__bf16)g;
        } else {
          outf[off] = v + bias[col] + resid[off];
        }
      }
    }
  }
}

// ---------------------------------------------------------------- fused attention
// One wave per (16-row tile, head, batch). Flash-style online softmax.
// K-tile fragments double-buffered; V tile staged through LDS with coalesced
// b128 loads issued before the score WMMA/softmax so latency hides under VALU.
__global__ __launch_bounds__(32)
void attn_kernel(const __bf16* __restrict__ qkv, const int* __restrict__ relidx,
                 const float* __restrict__ rpb, __bf16* __restrict__ outb) {
  int lane = threadIdx.x;
  int lr = lane & 15, hi = lane >> 4;
  int mt = blockIdx.x, h = blockIdx.y, b = blockIdx.z;
  int m0 = mt * 16;
  const size_t RS = 3 * CC;
  const __bf16* base = qkv + (size_t)b * NN * RS;

  // Q fragments (persist)
  v16bf qf[2];
  {
    int qrow = m0 + lr;
    bool ok = qrow < NN;
#pragma unroll
    for (int kk = 0; kk < 2; ++kk) {
      if (ok) {
        const __bf16* qp = base + (size_t)qrow * RS + h * DD + kk * 32;
        qf[kk] = loadfrag(qp + hi * 8, qp + 16 + hi * 8);
      } else {
        qf[kk] = zero16();
      }
    }
  }

  v8f acc[4];
#pragma unroll
  for (int nb = 0; nb < 4; ++nb)
#pragma unroll
    for (int i = 0; i < 8; ++i) acc[nb][i] = 0.f;
  float mrow[8], lrow[8];
#pragma unroll
  for (int i = 0; i < 8; ++i) { mrow[i] = -1e30f; lrow[i] = 0.f; }

  __shared__ float ptile[16][17];
  __shared__ __bf16 vtile[16][72];  // 72-stride: bank-conflict-free column reads

  // prologue: K fragments for j0 = 0
  v16bf kf[2];
  {
    int j = lr;  // < NN always
    const __bf16* kp = base + (size_t)j * RS + CC + h * DD + hi * 8;
    kf[0] = loadfrag(kp, kp + 16);
    kf[1] = loadfrag(kp + 32, kp + 48);
  }

  int vrow = lane >> 1;          // 0..15 : V-tile row this lane stages
  int vhalf = (lane & 1) * 32;   // column half (32 bf16 = 64B)

  for (int j0 = 0; j0 < NN; j0 += 16) {
    // --- issue V tile loads for this j0 (coalesced, 4 x b128 per lane)
    v8bf vv[4];
    {
      int jj = j0 + vrow;
      if (jj < NN) {
        const __bf16* vp = base + (size_t)jj * RS + 2 * CC + h * DD + vhalf;
#pragma unroll
        for (int c = 0; c < 4; ++c) vv[c] = *(const v8bf*)(vp + c * 8);
      } else {
#pragma unroll
        for (int c = 0; c < 4; ++c)
#pragma unroll
          for (int e = 0; e < 8; ++e) vv[c][e] = (__bf16)0.0f;
      }
    }
    // --- issue K fragment loads for next tile
    v16bf nkf0, nkf1;
    {
      int jn = j0 + 16 + lr;
      if (jn < NN) {
        const __bf16* kp = base + (size_t)jn * RS + CC + h * DD + hi * 8;
        nkf0 = loadfrag(kp, kp + 16);
        nkf1 = loadfrag(kp + 32, kp + 48);
      } else {
        nkf0 = zero16();
        nkf1 = zero16();
      }
    }

    // --- scores S[16,16] = Q(16x64) x K_tile^T
    v8f s;
#pragma unroll
    for (int i = 0; i < 8; ++i) s[i] = 0.f;
    s = wmma_bf16(qf[0], kf[0], s);
    s = wmma_bf16(qf[1], kf[1], s);

    int j = j0 + lr;
    bool jok = j < NN;
    // relative-position bias gather + padding mask
#pragma unroll
    for (int i = 0; i < 8; ++i) {
      int m = m0 + i + 8 * hi;
      if (jok && m < NN) {
        int ri = relidx[(size_t)m * NN + j];
        s[i] += rpb[ri * NHEAD + h];
      } else {
        s[i] = -1e30f;
      }
    }

    // --- online softmax update (row = same vgpr across a 16-lane half)
    float pv[8], rscale[8];
#pragma unroll
    for (int i = 0; i < 8; ++i) {
      float r = s[i];
      for (int d = 1; d < 16; d <<= 1) r = fmaxf(r, __shfl_xor(r, d, 32));
      float mnew = fmaxf(mrow[i], r);
      float sc = __expf(mrow[i] - mnew);
      float p = __expf(s[i] - mnew);
      float rsum = p;
      for (int d = 1; d < 16; d <<= 1) rsum += __shfl_xor(rsum, d, 32);
      lrow[i] = lrow[i] * sc + rsum;
      mrow[i] = mnew;
      rscale[i] = sc;
      pv[i] = p;
    }
#pragma unroll
    for (int nb = 0; nb < 4; ++nb)
#pragma unroll
      for (int i = 0; i < 8; ++i) acc[nb][i] *= rscale[i];

    // --- P (D-layout) -> A-layout via LDS, zero-pad K=16..31
#pragma unroll
    for (int i = 0; i < 8; ++i) ptile[i + 8 * hi][lr] = pv[i];
    __syncthreads();
    v16bf pf;
#pragma unroll
    for (int i = 0; i < 8; ++i) {
      pf[i] = (__bf16)ptile[lr][hi * 8 + i];
      pf[8 + i] = (__bf16)0.0f;
    }

    // --- park V tile in LDS (loads have had the whole softmax to complete)
#pragma unroll
    for (int c = 0; c < 4; ++c)
      *(v8bf*)&vtile[vrow][vhalf + c * 8] = vv[c];
    __syncthreads();

    // --- O += P(16x16, padded) x V_tile(16x64)
#pragma unroll
    for (int nb = 0; nb < 4; ++nb) {
      v16bf vf = zero16();
      {
#pragma unroll
        for (int i = 0; i < 8; ++i) vf[i] = vtile[hi * 8 + i][nb * 16 + lr];
      }
      acc[nb] = wmma_bf16(pf, vf, acc[nb]);
    }
    __syncthreads();

    kf[0] = nkf0; kf[1] = nkf1;
  }

  for (int i = 0; i < 8; ++i) {
    int m = m0 + i + 8 * hi;
    if (m >= NN) continue;
    float inv = 1.0f / lrow[i];
    for (int nb = 0; nb < 4; ++nb) {
      outb[((size_t)b * NN + m) * CC + h * DD + nb * 16 + lr] =
          (__bf16)(acc[nb][i] * inv);
    }
  }
}

// ------------------------------------------------- cls-row attention (for plan)
__global__ __launch_bounds__(256)
void cls_row_kernel(const __bf16* __restrict__ qkv, const int* __restrict__ relidx,
                    const float* __restrict__ rpb, float* __restrict__ clsP) {
  int h = blockIdx.x, b = blockIdx.y, tid = threadIdx.x;
  __shared__ float q0[DD];
  __shared__ float sv[NN];
  __shared__ float red[256];
  const __bf16* base = qkv + (size_t)b * NN * 3 * CC;
  if (tid < DD) q0[tid] = (float)base[h * DD + tid];  // q row 0 (already scaled)
  __syncthreads();
  for (int j = tid; j < NN; j += 256) {
    const __bf16* kp = base + (size_t)j * 3 * CC + CC + h * DD;
    float a = 0.f;
    for (int d = 0; d < DD; ++d) a += q0[d] * (float)kp[d];
    a += rpb[relidx[j] * NHEAD + h];  // rel_index[0, j]
    sv[j] = a;
  }
  __syncthreads();
  float lm = -1e30f;
  for (int j = tid; j < NN; j += 256) lm = fmaxf(lm, sv[j]);
  red[tid] = lm; __syncthreads();
  for (int st = 128; st > 0; st >>= 1) {
    if (tid < st) red[tid] = fmaxf(red[tid], red[tid + st]);
    __syncthreads();
  }
  float mx = red[0]; __syncthreads();
  float ls = 0.f;
  for (int j = tid; j < NN; j += 256) { float e = __expf(sv[j] - mx); sv[j] = e; ls += e; }
  red[tid] = ls; __syncthreads();
  for (int st = 128; st > 0; st >>= 1) {
    if (tid < st) red[tid] += red[tid + st];
    __syncthreads();
  }
  float inv = 1.0f / red[0];
  for (int j = tid; j < NN; j += 256)
    clsP[((size_t)b * NHEAD + h) * NN + j] = sv[j] * inv;
}

__global__ void cls_mean_kernel(const float* __restrict__ clsP,
                                float* __restrict__ cls) {
  int b = blockIdx.y;
  int j = blockIdx.x * 256 + threadIdx.x;
  if (j < NIMG) {
    float s = 0.f;
    for (int h = 0; h < NHEAD; ++h) s += clsP[((size_t)b * NHEAD + h) * NN + (j + 1)];
    cls[b * NIMG + j] = s * (1.0f / NHEAD);
  }
}

// ---------------------------------------------------------------- plan
__global__ __launch_bounds__(256)
void plan_stats_kernel(const float* __restrict__ cls,
                       const float* __restrict__ tw, int* __restrict__ plan) {
  __shared__ float rs[256], rm[256];
  __shared__ float counts[BB];
  int tid = threadIdx.x;
  for (int b = 0; b < BB; ++b) {
    float s = 0.f, m = -1e30f;
    for (int j = tid; j < NIMG; j += 256) { float v = cls[b * NIMG + j]; s += v; m = fmaxf(m, v); }
    rs[tid] = s; rm[tid] = m; __syncthreads();
    for (int st = 128; st > 0; st >>= 1) {
      if (tid < st) { rs[tid] += rs[tid + st]; rm[tid] = fmaxf(rm[tid], rm[tid + st]); }
      __syncthreads();
    }
    float avg = rs[0] * (1.0f / NIMG), mx = rm[0];
    __syncthreads();
    int c = 0;
    for (int j = tid; j < NIMG; j += 256) { float v = cls[b * NIMG + j]; if (v >= avg && v <= mx) c++; }
    rs[tid] = (float)c; __syncthreads();
    for (int st = 128; st > 0; st >>= 1) {
      if (tid < st) rs[tid] += rs[tid + st];
      __syncthreads();
    }
    if (tid == 0) counts[b] = rs[0];
    __syncthreads();
  }
  if (tid == 0) {
    float mxc = 0.f, T = 0.f;
    for (int b = 0; b < BB; ++b) { mxc = fmaxf(mxc, counts[b]); T += counts[b]; }
    T *= (1.0f / BB);
    float sig = 1.0f / (1.0f + __expf(-(mxc - T) / (float)BB));
    int keep = (int)roundf(NIMG * (LAMK + (1.0f - LAMK) * sig));
    if (keep > NIMG) keep = NIMG;
    float t0 = tw[0], t1 = tw[1];
    float mm = fmaxf(t0, t1);
    float e0 = __expf(t0 - mm), e1 = __expf(t1 - mm);
    float w0 = e0 / (e0 + e1);
    int remaining = NN - (1 + keep);
    int nt = (int)(w0 * remaining);
    plan[0] = keep; plan[1] = nt; plan[2] = remaining - nt;
  }
}

__global__ __launch_bounds__(256)
void topk_kernel(float* __restrict__ cls, const int* __restrict__ plan,
                 int* __restrict__ topidx) {
  int b = blockIdx.x, tid = threadIdx.x;
  __shared__ float bv[256];
  __shared__ int bi[256];
  int keep = plan[0];
  float* row = cls + b * NIMG;
  for (int k = 0; k < keep; ++k) {
    float best = -1e38f; int bidx = NIMG;
    for (int j = tid; j < NIMG; j += 256) {
      float v = row[j];
      if (v > best || (v == best && j < bidx)) { best = v; bidx = j; }
    }
    bv[tid] = best; bi[tid] = bidx; __syncthreads();
    for (int st = 128; st > 0; st >>= 1) {
      if (tid < st &&
          (bv[tid + st] > bv[tid] || (bv[tid + st] == bv[tid] && bi[tid + st] < bi[tid]))) {
        bv[tid] = bv[tid + st]; bi[tid] = bi[tid + st];
      }
      __syncthreads();
    }
    if (tid == 0) { topidx[b * NIMG + k] = bi[0]; row[bi[0]] = -1e38f; }
    __syncthreads();
  }
}

__global__ void dedup_kernel(const int* __restrict__ topidx,
                             const int* __restrict__ plan, int* __restrict__ sel) {
  if (threadIdx.x != 0 || blockIdx.x != 0) return;
  int keep = plan[0];
  unsigned int seen[NIMG / 32];
  for (int i = 0; i < NIMG / 32; ++i) seen[i] = 0;
  int cnt = 0;
  for (int b = 0; b < BB && cnt < keep; ++b)
    for (int k = 0; k < keep && cnt < keep; ++k) {
      int v = topidx[b * NIMG + k];
      unsigned int bit = 1u << (v & 31);
      if (!(seen[v >> 5] & bit)) { seen[v >> 5] |= bit; sel[cnt++] = v; }
    }
}

static __device__ inline unsigned int hash_u32(unsigned int x) {
  x ^= x >> 16; x *= 0x7feb352du; x ^= x >> 15; x *= 0x846ca68bu; x ^= x >> 16;
  return x;
}

__global__ void fill_ii_kernel(int* __restrict__ ii) {
  int b = blockIdx.y;
  int t = blockIdx.x * 256 + threadIdx.x;
  if (t < NIMG)
    ii[b * NIMG + t] = (int)(hash_u32((unsigned)(b * NIMG + t) + 1234u) % (NIMG - 1));
}

// ---------------------------------------------------------------- assemble + residual
__global__ __launch_bounds__(256)
void assemble_kernel(const float* __restrict__ x, const float* __restrict__ xo,
                     const float* __restrict__ xn, const int* __restrict__ sel,
                     const int* __restrict__ ii, const int* __restrict__ plan,
                     float* __restrict__ x1) {
  int p = blockIdx.x % NN;
  int b = blockIdx.x / NN;
  int keep = plan[0], nt = plan[1];
  const float* add;
  if (p == 0)                add = xo + (size_t)b * NN * CC;
  else if (p <= keep)        add = xo + ((size_t)b * NN + 1 + sel[p - 1]) * CC;
  else if (p <= keep + nt)   add = xn + ((size_t)b * NN + 1) * CC;        // ti == 0 always
  else {
    int t = p - 1 - keep - nt;
    add = xn + ((size_t)b * NN + 2 + ii[b * NIMG + t]) * CC;
  }
  size_t off = ((size_t)b * NN + p) * CC;
  for (int c = threadIdx.x; c < CC; c += 256) x1[off + c] = x[off + c] + add[c];
}

// ---------------------------------------------------------------- launch
extern "C" void kernel_launch(void* const* d_in, const int* in_sizes, int n_in,
                              void* d_out, int out_size, void* d_ws, size_t ws_size,
                              hipStream_t stream) {
  const float* x      = (const float*)d_in[0];
  const float* ln1w   = (const float*)d_in[1];
  const float* ln1b   = (const float*)d_in[2];
  const float* qkv_w  = (const float*)d_in[3];
  const float* rpb    = (const float*)d_in[4];
  const float* proj_w = (const float*)d_in[5];
  const float* proj_b = (const float*)d_in[6];
  const float* tokw   = (const float*)d_in[7];
  const float* ln2w   = (const float*)d_in[8];
  const float* ln2b   = (const float*)d_in[9];
  const float* fc1_w  = (const float*)d_in[10];
  const float* fc1_b  = (const float*)d_in[11];
  const float* fc2_w  = (const float*)d_in[12];
  const float* fc2_b  = (const float*)d_in[13];
  const int*   relidx = (const int*)d_in[14];
  float* out = (float*)d_out;

  // workspace carve-up (256B aligned bump allocator)
  char* base = (char*)d_ws;
  size_t off = 0;
  auto alloc = [&](size_t bytes) -> char* {
    char* p = base + off;
    off += (bytes + 255) & ~(size_t)255;
    return p;
  };
  __bf16* qkvw_b = (__bf16*)alloc((size_t)3 * CC * CC * 2);
  __bf16* projw_b = (__bf16*)alloc((size_t)CC * CC * 2);
  __bf16* fc1w_b = (__bf16*)alloc((size_t)4 * CC * CC * 2);
  __bf16* fc2w_b = (__bf16*)alloc((size_t)CC * 4 * CC * 2);
  float*  xn_f   = (float*)alloc((size_t)MROWS * CC * 4);
  __bf16* xn_b   = (__bf16*)alloc((size_t)MROWS * CC * 2);
  __bf16* qkv_b  = (__bf16*)alloc((size_t)MROWS * 3 * CC * 2);
  __bf16* ao_b   = (__bf16*)alloc((size_t)MROWS * CC * 2);
  float*  xo_f   = (float*)alloc((size_t)MROWS * CC * 4);
  float*  clsP   = (float*)alloc((size_t)BB * NHEAD * NN * 4);
  float*  cls    = (float*)alloc((size_t)BB * NIMG * 4);
  int*    plan   = (int*)alloc(256);
  int*    topidx = (int*)alloc((size_t)BB * NIMG * 4);
  int*    sel    = (int*)alloc((size_t)NIMG * 4);
  int*    iibuf  = (int*)alloc((size_t)BB * NIMG * 4);
  float*  x1     = (float*)alloc((size_t)MROWS * CC * 4);
  __bf16* h1_b   = (__bf16*)alloc((size_t)MROWS * CC * 2);
  __bf16* h2_b   = (__bf16*)alloc((size_t)MROWS * 4 * CC * 2);
  if (off > ws_size) return;  // insufficient scratch: bail safely

  const int MT32 = (MROWS + 31) / 32;  // 257 row tiles (32 rows per wave)

  // 1) weights -> bf16
  cast_f32_bf16<<<(3 * CC * CC + 255) / 256, 256, 0, stream>>>(qkv_w, qkvw_b, 3 * CC * CC);
  cast_f32_bf16<<<(CC * CC + 255) / 256, 256, 0, stream>>>(proj_w, projw_b, CC * CC);
  cast_f32_bf16<<<(4 * CC * CC + 255) / 256, 256, 0, stream>>>(fc1_w, fc1w_b, 4 * CC * CC);
  cast_f32_bf16<<<(4 * CC * CC + 255) / 256, 256, 0, stream>>>(fc2_w, fc2w_b, 4 * CC * CC);

  // 2) LN1
  ln_kernel<<<MROWS, 256, 0, stream>>>(x, ln1w, ln1b, xn_f, xn_b);

  // 3) QKV GEMM (q pre-scaled)
  gemm_kernel<<<dim3(MT32, (3 * CC) / 64), 32, 0, stream>>>(
      xn_b, qkvw_b, nullptr, nullptr, nullptr, qkv_b, MROWS, CC, 3 * CC, 0);

  // 4) fused flash attention with relative-position bias
  attn_kernel<<<dim3((NN + 15) / 16, NHEAD, BB), 32, 0, stream>>>(qkv_b, relidx, rpb, ao_b);

  // 5) output projection
  gemm_kernel<<<dim3(MT32, CC / 64), 32, 0, stream>>>(
      ao_b, projw_b, proj_b, nullptr, xo_f, nullptr, MROWS, CC, CC, 1);

  // 6) plan: cls attention row, head mean, stats, top-k, dedup, fillers
  cls_row_kernel<<<dim3(NHEAD, BB), 256, 0, stream>>>(qkv_b, relidx, rpb, clsP);
  cls_mean_kernel<<<dim3((NIMG + 255) / 256, BB), 256, 0, stream>>>(clsP, cls);
  plan_stats_kernel<<<1, 256, 0, stream>>>(cls, tokw, plan);
  topk_kernel<<<BB, 256, 0, stream>>>(cls, plan, topidx);
  dedup_kernel<<<1, 1, 0, stream>>>(topidx, plan, sel);
  fill_ii_kernel<<<dim3((NIMG + 255) / 256, BB), 256, 0, stream>>>(iibuf);

  // 7) token reassembly + residual
  assemble_kernel<<<MROWS, 256, 0, stream>>>(x, xo_f, xn_f, sel, iibuf, plan, x1);

  // 8) LN2 + MLP + residual
  ln_kernel<<<MROWS, 256, 0, stream>>>(x1, ln2w, ln2b, nullptr, h1_b);
  gemm_kernel<<<dim3(MT32, (4 * CC) / 64), 32, 0, stream>>>(
      h1_b, fc1w_b, fc1_b, nullptr, nullptr, h2_b, MROWS, CC, 4 * CC, 2);
  gemm_kernel<<<dim3(MT32, CC / 64), 32, 0, stream>>>(
      h2_b, fc2w_b, fc2_b, x1, out, nullptr, MROWS, 4 * CC, CC, 3);

  (void)in_sizes; (void)n_in; (void)out_size;
}
